// GNN_1520418423297
// MI455X (gfx1250) — matrix operand order repaired and checked by reading, not compiled, not run on hardware
//
#include <hip/hip_runtime.h>

// GCN 2-layer forward for MI455X (gfx1250, wave32).
// Dense transforms: bf16 WMMA (v_wmma_f32_16x16x32_bf16), fp32 accumulate,
//   one wave per 16x64 output strip (4 accumulators, A-fragment reused 4x),
//   K fully unrolled -> 16 straight-line WMMAs per wave.
// Message passing: float4 gathers + global f32 atomics (all activations are
//   L2-resident: h1/agg1 = 25.6 MB each << 192 MB L2).

typedef __attribute__((ext_vector_type(16))) __bf16 v16bf;
typedef __attribute__((ext_vector_type(8)))  float  v8f;

#define NNODES 50000
#define NEDGES 800000
#define FIN    128
#define HID0   128
#define HID1   64

// ---------------- degree / norm ----------------

__global__ void k_init_deg(float* __restrict__ deg, int n) {
    int i = blockIdx.x * blockDim.x + threadIdx.x;
    if (i < n) deg[i] = 1.0f;               // self-loop contributes 1
}

__global__ void k_count_deg(const int* __restrict__ edges, float* __restrict__ deg, int e) {
    int i = blockIdx.x * blockDim.x + threadIdx.x;
    if (i < e) atomicAdd(&deg[edges[2 * i + 1]], 1.0f);
}

__global__ void k_rsqrt_inplace(float* __restrict__ d, int n) {
    int i = blockIdx.x * blockDim.x + threadIdx.x;
    if (i < n) d[i] = rsqrtf(d[i]);
}

__global__ void k_zero(float* __restrict__ p, int n) {
    int i = blockIdx.x * blockDim.x + threadIdx.x;
    if (i < n) p[i] = 0.0f;
}

// ---------------- WMMA GEMM: C[M x NC] = A[M x K] * B[K x NC] ----------------
// fp32 memory, bf16 compute, fp32 accumulate.
// One wave (32 threads) per 16-row x 64-col strip: 4 v8f accumulators, the A
// fragment is loaded/converted once per 32-K step and feeds 4 WMMAs.
// gridDim = (M/16, NC/64). Requires M%16==0, K%32==0, NC%64==0.

template <int K, int NC>
__global__ void __launch_bounds__(32)
k_gemm_bf16(const float* __restrict__ A, const float* __restrict__ B,
            float* __restrict__ C) {
    const int lane    = threadIdx.x & 31;
    const int row0    = blockIdx.x * 16;
    const int colbase = blockIdx.y * 64;
    const int half    = lane >> 4;    // 0: lanes 0-15, 1: lanes 16-31
    const int l15     = lane & 15;
    const int abase   = half * 16;    // A fragment K-offset within group of 32
    const int bkoff   = half * 16;    // B fragment K-offset within group of 32

    v8f acc0 = {}, acc1 = {}, acc2 = {}, acc3 = {};
    const float* __restrict__ arow = A + (size_t)(row0 + l15) * K;

#pragma unroll
    for (int kk = 0; kk < K; kk += 32) {
        // A fragment: lane holds row (row0+l15); lanes 0-15 K=kk..kk+15, lanes 16-31 K=kk+16..kk+31.
        v16bf a;
#pragma unroll
        for (int j = 0; j < 8; ++j)
            a[j] = (__bf16)arow[kk + abase + j];
#pragma unroll
        for (int j = 0; j < 8; ++j)
            a[8 + j] = (__bf16)arow[kk + abase + 8 + j];

        // B fragments for 4 column tiles: lane holds column (colbase+l15+16*t);
        // lanes 0-15 cover K=kk..kk+15, lanes 16-31 K=kk+16..kk+31.
        const float* __restrict__ bp =
            B + (size_t)(kk + bkoff) * NC + colbase + l15;
        v16bf b0, b1, b2, b3;
#pragma unroll
        for (int j = 0; j < 16; ++j) {
            const float* __restrict__ r = bp + (size_t)j * NC;
            b0[j] = (__bf16)r[0];
            b1[j] = (__bf16)r[16];
            b2[j] = (__bf16)r[32];
            b3[j] = (__bf16)r[48];
        }

        acc0 = __builtin_amdgcn_wmma_f32_16x16x32_bf16(false, a, false, b0, (short)0, acc0, false, false);
        acc1 = __builtin_amdgcn_wmma_f32_16x16x32_bf16(false, a, false, b1, (short)0, acc1, false, false);
        acc2 = __builtin_amdgcn_wmma_f32_16x16x32_bf16(false, a, false, b2, (short)0, acc2, false, false);
        acc3 = __builtin_amdgcn_wmma_f32_16x16x32_bf16(false, a, false, b3, (short)0, acc3, false, false);
    }

    // C strip: VGPR r on lane l -> row row0 + r + 8*half, col colbase + l15 + 16*t.
    float* __restrict__ crow = C + (size_t)(row0 + half * 8) * NC + colbase + l15;
#pragma unroll
    for (int r = 0; r < 8; ++r) {
        float* __restrict__ cr = crow + (size_t)r * NC;
        cr[0]  = acc0[r];
        cr[16] = acc1[r];
        cr[32] = acc2[r];
        cr[48] = acc3[r];
    }
}

// ---------------- edge scatter: agg[dst] += h[src] * dinv[src]*dinv[dst] ----------------
// F/4 threads per edge, each moves 4 features (float4 gather + 4 f32 atomics).

template <int F>
__global__ void k_scatter(const int* __restrict__ edges,
                          const float* __restrict__ dinv,
                          const float* __restrict__ h,
                          float* __restrict__ agg, int e) {
    constexpr int TPE = F / 4;
    long long t = (long long)blockIdx.x * blockDim.x + threadIdx.x;
    int eid  = (int)(t / TPE);
    int part = (int)(t % TPE);
    if (eid >= e) return;
    int s = edges[2 * eid + 0];
    int d = edges[2 * eid + 1];
    float w = dinv[s] * dinv[d];
    int f0 = part * 4;
    const float4 hv = *(const float4*)(h + (size_t)s * F + f0);
    float* __restrict__ o = agg + (size_t)d * F + f0;
    atomicAdd(o + 0, hv.x * w);
    atomicAdd(o + 1, hv.y * w);
    atomicAdd(o + 2, hv.z * w);
    atomicAdd(o + 3, hv.w * w);
}

// ---------------- epilogue: self-loop + bias (+ReLU) ----------------
// out[i,f] = act( agg[i,f] + h[i,f]*dinv[i]^2 + bias[f] )

template <bool RELU, int F>
__global__ void k_post(const float* __restrict__ agg,
                       const float* __restrict__ h,
                       const float* __restrict__ dinv,
                       const float* __restrict__ bias,
                       float* __restrict__ out, int n) {
    int t = blockIdx.x * blockDim.x + threadIdx.x;
    if (t >= n * F) return;
    int i = t / F;
    int f = t % F;
    float di = dinv[i];
    float v = agg[t] + h[t] * di * di + bias[f];
    if (RELU) v = fmaxf(v, 0.0f);
    out[t] = v;
}

// ---------------- launch ----------------

extern "C" void kernel_launch(void* const* d_in, const int* in_sizes, int n_in,
                              void* d_out, int out_size, void* d_ws, size_t ws_size,
                              hipStream_t stream) {
    const float* X     = (const float*)d_in[0];
    const int*   edges = (const int*)d_in[1];   // [E,2] src,dst
    const float* W1    = (const float*)d_in[2];
    const float* b1    = (const float*)d_in[3];
    const float* W2    = (const float*)d_in[4];
    const float* b2    = (const float*)d_in[5];
    float* out = (float*)d_out;

    // Workspace layout (floats):
    //  [0, 50048)                      dinv (first used as deg)
    //  [50048, 50048 + N*128)          h1   (layer 2 reuses: h2 = first N*64, agg2 = next N*64)
    //  [.. + N*128, .. + 2*N*128)      agg1 -> x1 (in place)
    float* ws   = (float*)d_ws;
    float* dinv = ws;
    float* h1   = ws + 50048;
    float* agg1 = h1 + (size_t)NNODES * FIN;
    float* h2   = h1;                            // N*H1
    float* agg2 = h1 + (size_t)NNODES * HID1;    // N*H1

    const int T = 256;

    // degrees & symmetric norm
    k_init_deg<<<(NNODES + T - 1) / T, T, 0, stream>>>(dinv, NNODES);
    k_count_deg<<<(NEDGES + T - 1) / T, T, 0, stream>>>(edges, dinv, NEDGES);
    k_rsqrt_inplace<<<(NNODES + T - 1) / T, T, 0, stream>>>(dinv, NNODES);

    // ---- layer 1: h1 = X @ W1 ; agg1 = scatter(h1) ; x1 = relu(agg1 + self + b1)
    k_zero<<<(NNODES * FIN + T - 1) / T, T, 0, stream>>>(agg1, NNODES * FIN);
    k_gemm_bf16<FIN, HID0><<<dim3(NNODES / 16, HID0 / 64), 32, 0, stream>>>(X, W1, h1);
    {
        long long nthr = (long long)NEDGES * (FIN / 4);
        k_scatter<FIN><<<(int)((nthr + T - 1) / T), T, 0, stream>>>(edges, dinv, h1, agg1, NEDGES);
    }
    k_post<true, FIN><<<(NNODES * FIN + T - 1) / T, T, 0, stream>>>(
        agg1, h1, dinv, b1, agg1, NNODES);   // x1 in place of agg1

    // ---- layer 2: h2 = x1 @ W2 ; agg2 = scatter(h2) ; out = agg2 + self + b2
    k_gemm_bf16<HID0, HID1><<<dim3(NNODES / 16, HID1 / 64), 32, 0, stream>>>(agg1, W2, h2);
    k_zero<<<(NNODES * HID1 + T - 1) / T, T, 0, stream>>>(agg2, NNODES * HID1);
    {
        long long nthr = (long long)NEDGES * (HID1 / 4);
        k_scatter<HID1><<<(int)((nthr + T - 1) / T), T, 0, stream>>>(edges, dinv, h2, agg2, NEDGES);
    }
    k_post<false, HID1><<<(NNODES * HID1 + T - 1) / T, T, 0, stream>>>(
        agg2, h2, dinv, b2, out, NNODES);
}
